// SwinTransformerBlock_5918464934075
// MI455X (gfx1250) — compile-verified, matrix operand order checked by reading, not run on hardware
//
#include <hip/hip_runtime.h>
#include <hip/hip_bf16.h>

typedef __attribute__((ext_vector_type(16))) _Float16 v16h;
typedef __attribute__((ext_vector_type(8)))  _Float16 v8h;
typedef __attribute__((ext_vector_type(8)))  float    v8f;

#define HD    32      // head dim
#define NTOK  49      // tokens per window
#define NPAD  64      // padded tokens per window
#define SDIM  192     // model dim
#define HEADS 6
#define MTOT  100352  // B*H*W tokens
#define MLPD  768
#define NWH   12288   // (B*nW)*HEADS

// ---------------------------------------------------------------------------
// Fragment loader for v_wmma_f32_16x16x32_f16.
// 16-bit A/B fragment layout (ISA 7.12.2): lane%16 selects the M (or N) row,
// the 16 f16 elements are K = {base..base+7, base+16..base+23}, base = 8*(lane>=16).
// With row-major [row][K] data (K-offset 16B aligned) that is two b128 loads.
// ---------------------------------------------------------------------------
__device__ __forceinline__ v16h load_frag(const _Float16* p) {
  v8h lo = *(const v8h*)p;
  v8h hi = *(const v8h*)(p + 16);
  v16h r;
#pragma unroll
  for (int e = 0; e < 8; ++e) { r[e] = lo[e]; r[e + 8] = hi[e]; }
  return r;
}

__device__ __forceinline__ v8f wmma_f16(v16h a, v16h b, v8f c) {
  return __builtin_amdgcn_wmma_f32_16x16x32_f16(false, a, false, b, (short)0, c,
                                                false, false);
}

// zone id for the Swin shift mask (slices [0,49),[49,53),[53,56))
__device__ __forceinline__ int swin_zone(int g) { return g < 49 ? 0 : (g < 53 ? 1 : 2); }

// ---------------------------------------------------------------------------
// Weight convert: wt[n*K + k] = (f16) w[k*N + n]   (pre-transposed for B frags)
// ---------------------------------------------------------------------------
__global__ void cvt_wT(const float* __restrict__ w, _Float16* __restrict__ wt,
                       int K, int N) {
  int i = blockIdx.x * 256 + threadIdx.x;
  if (i < K * N) {
    int n = i / K, kk = i - n * K;
    wt[i] = (_Float16)w[(size_t)kk * N + n];
  }
}

// ---------------------------------------------------------------------------
// One-time attention tables (keeps the softmax hot loop branch-free):
//   bias_tab[h][nq][nk]  = relative-position bias (0 outside 49x49)
//   mask_tab[wl][nq][nk] = shift mask (0 / -100), -1e30 on padded rows/cols
// ---------------------------------------------------------------------------
__global__ void build_tabs(const float* __restrict__ rel_bias,
                           float* __restrict__ bias_tab,
                           float* __restrict__ mask_tab) {
  int i = blockIdx.x * 256 + threadIdx.x;  // 0..4095 -> (nq, nk)
  int nq = i >> 6, nk = i & 63;
  bool valid = (nq < NTOK) && (nk < NTOK);
  int qr = nq / 7, qc = nq - qr * 7;
  int kr = nk / 7, kc = nk - kr * 7;
#pragma unroll
  for (int h = 0; h < HEADS; ++h) {
    float bv = 0.f;
    if (valid) bv = rel_bias[((qr - kr + 6) * 13 + (qc - kc + 6)) * HEADS + h];
    bias_tab[h * 4096 + i] = bv;
  }
  for (int wl = 0; wl < 64; ++wl) {
    int wr = wl >> 3, wc = wl & 7;
    float mv;
    if (!valid) {
      mv = -1e30f;
    } else {
      int zq = swin_zone(wr * 7 + qr) * 3 + swin_zone(wc * 7 + qc);
      int zk = swin_zone(wr * 7 + kr) * 3 + swin_zone(wc * 7 + kc);
      mv = (zq != zk) ? -100.0f : 0.0f;
    }
    mask_tab[wl * 4096 + i] = mv;
  }
}

// ---------------------------------------------------------------------------
// LayerNorm, one wave per token (wave32). REMAP=true also applies the cyclic
// shift + window partition so the output rows are in window order.
// ---------------------------------------------------------------------------
template <bool REMAP>
__global__ __launch_bounds__(256) void ln_kernel(const float* __restrict__ x,
                                                 const float* __restrict__ g,
                                                 const float* __restrict__ b,
                                                 _Float16* __restrict__ out) {
  int lane = threadIdx.x & 31;
  int t = blockIdx.x * 8 + (threadIdx.x >> 5);  // token row (window order)
  const float* xp;
  if constexpr (REMAP) {
    int win = t / NTOK, tok = t - win * NTOK;
    int b_ = win >> 6, wl = win & 63;
    int wr = wl >> 3, wc = wl & 7;
    int tr = tok / 7, tc = tok - tr * 7;
    int r = (wr * 7 + tr + 3) % 56, c = (wc * 7 + tc + 3) % 56;  // roll(-3)
    xp = x + ((size_t)b_ * 3136 + r * 56 + c) * SDIM;
  } else {
    xp = x + (size_t)t * SDIM;
  }
  float v[6], s = 0.f;
#pragma unroll
  for (int j = 0; j < 6; ++j) { v[j] = xp[lane + 32 * j]; s += v[j]; }
#pragma unroll
  for (int m = 16; m; m >>= 1) s += __shfl_xor(s, m);
  float mu = s * (1.0f / 192.0f);
  float var = 0.f;
#pragma unroll
  for (int j = 0; j < 6; ++j) { float d = v[j] - mu; var += d * d; }
#pragma unroll
  for (int m = 16; m; m >>= 1) var += __shfl_xor(var, m);
  float inv = rsqrtf(var * (1.0f / 192.0f) + 1e-5f);
#pragma unroll
  for (int j = 0; j < 6; ++j) {
    int ci = lane + 32 * j;
    out[(size_t)t * SDIM + ci] = (_Float16)((v[j] - mu) * inv * g[ci] + b[ci]);
  }
}

// ---------------------------------------------------------------------------
// WMMA GEMM: C[M][N] = A[M][K] (f16) * Bt[N][K]^T (f16) + bias, fp32 acc.
// Block = 256 threads = 8 waves covering a 128(M) x 64(N) tile; each wave owns
// a 32x32 register tile (2x2 fragments, 4 WMMAs per 32-wide K step).
// MODE: 0=QKV scatter, 1=proj+reverse+residual, 2=fc1+GELU, 3=fc2+residual.
// ---------------------------------------------------------------------------
template <int MODE>
__global__ __launch_bounds__(256) void gemm_wmma(
    const _Float16* __restrict__ A, const _Float16* __restrict__ Bt,
    const float* __restrict__ bias, int K,
    _Float16* __restrict__ out_h, float* __restrict__ out_f,
    const float* __restrict__ resid,
    _Float16* __restrict__ qout, _Float16* __restrict__ kout,
    _Float16* __restrict__ vout) {
  int lane = threadIdx.x & 31, wv = threadIdx.x >> 5;
  int wm = wv >> 1, wn = wv & 1;
  int mBase = blockIdx.x * 128 + wm * 32;
  int nBase = blockIdx.y * 64 + wn * 32;
  int lrow = lane & 15, hi8 = (lane >> 4) * 8;

  v8f acc[2][2] = {};
  for (int k0 = 0; k0 < K; k0 += 32) {
    v16h a[2], b[2];
#pragma unroll
    for (int i = 0; i < 2; ++i) {
      const _Float16* ap = A + (size_t)(mBase + i * 16 + lrow) * K + k0 + hi8;
      __builtin_prefetch(ap + 32, 0, 1);  // next K-step
      a[i] = load_frag(ap);
    }
#pragma unroll
    for (int j = 0; j < 2; ++j)
      b[j] = load_frag(Bt + (size_t)(nBase + j * 16 + lrow) * K + k0 + hi8);
#pragma unroll
    for (int i = 0; i < 2; ++i)
#pragma unroll
      for (int j = 0; j < 2; ++j) acc[i][j] = wmma_f16(a[i], b[j], acc[i][j]);
  }

#pragma unroll
  for (int i = 0; i < 2; ++i)
#pragma unroll
    for (int j = 0; j < 2; ++j)
#pragma unroll
      for (int e = 0; e < 8; ++e) {
        // C-fragment layout: M = e + 8*(lane>=16), N = lane%16 (per 16x16 tile)
        int M = mBase + i * 16 + hi8 + e;
        int Nc = nBase + j * 16 + lrow;
        float v = acc[i][j][e] + bias[Nc];
        if constexpr (MODE == 0) {  // QKV scatter into per-(win,head) buffers
          int which = Nc / SDIM, rr = Nc % SDIM;
          int hh = rr >> 5, d = rr & 31;
          int win = M / NTOK, tok = M - win * NTOK;
          size_t wh = (size_t)win * HEADS + hh;
          if (which == 0)
            qout[(wh * NPAD + tok) * HD + d] = (_Float16)(v * 0.17677669529663689f);
          else if (which == 1)
            kout[(wh * NPAD + tok) * HD + d] = (_Float16)v;
          else
            vout[(wh * HD + d) * NPAD + tok] = (_Float16)v;  // V transposed
        } else if constexpr (MODE == 1) {  // proj + window reverse + unshift + add x
          int win = M / NTOK, tok = M - win * NTOK;
          int b_ = win >> 6, wl = win & 63;
          int wr = wl >> 3, wc = wl & 7;
          int tr = tok / 7, tc = tok - tr * 7;
          int r = (wr * 7 + tr + 3) % 56, c = (wc * 7 + tc + 3) % 56;  // roll(+3)
          size_t pos = ((size_t)b_ * 3136 + r * 56 + c) * SDIM + Nc;
          out_f[pos] = v + resid[pos];
        } else if constexpr (MODE == 2) {  // fc1 + exact GELU
          float ge = 0.5f * v * (1.0f + erff(v * 0.70710678118654752f));
          out_h[(size_t)M * MLPD + Nc] = (_Float16)ge;
        } else {  // fc2 + residual -> final output
          size_t pos = (size_t)M * SDIM + Nc;
          out_f[pos] = v + resid[pos];
        }
      }
}

// ---------------------------------------------------------------------------
// Windowed attention: one wave per (window, head). 49x49 padded to 64x64.
// S = Q*K^T (4x4 WMMA tiles); bias+mask come from precomputed L2-resident
// tables (branch-free); softmax in the C-fragment layout via shfl over the
// 16-lane halves; P restriped through LDS; O = P*V^T (4 WMMAs).
// ---------------------------------------------------------------------------
__global__ __launch_bounds__(256) void attn_kernel(
    const _Float16* __restrict__ q, const _Float16* __restrict__ k,
    const _Float16* __restrict__ vT, const float* __restrict__ bias_tab,
    const float* __restrict__ mask_tab, _Float16* __restrict__ out) {
  __shared__ _Float16 ldsP[8][16 * 64];
  int lane = threadIdx.x & 31, wv = threadIdx.x >> 5;
  int wh = blockIdx.x * 8 + wv;
  int win = wh / HEADS, head = wh - win * HEADS;
  const _Float16* qb = q + (size_t)wh * NPAD * HD;
  const _Float16* kb = k + (size_t)wh * NPAD * HD;
  const _Float16* vb = vT + (size_t)wh * HD * NPAD;
  int lrow = lane & 15, hi8 = (lane >> 4) * 8;
  const float* btab = bias_tab + head * 4096 + lrow;
  const float* mtab = mask_tab + (win & 63) * 4096 + lrow;

  for (int mt = 0; mt < 4; ++mt) {
    int m0 = mt * 16;
    v16h aq = load_frag(qb + (size_t)(m0 + lrow) * HD + hi8);
    v8f S[4];
#pragma unroll
    for (int nt = 0; nt < 4; ++nt) {
      v16h bk = load_frag(kb + (size_t)(nt * 16 + lrow) * HD + hi8);
      v8f z = {};
      S[nt] = wmma_f16(aq, bk, z);
    }
    // add bias+mask, then softmax; each S row lives in one 16-lane half
#pragma unroll
    for (int e = 0; e < 8; ++e) {
      int nq = m0 + hi8 + e;
      const float* brow = btab + nq * 64;
      const float* mrow = mtab + nq * 64;
      float vals[4];
#pragma unroll
      for (int nt = 0; nt < 4; ++nt)
        vals[nt] = S[nt][e] + brow[nt * 16] + mrow[nt * 16];
      float m = fmaxf(fmaxf(vals[0], vals[1]), fmaxf(vals[2], vals[3]));
#pragma unroll
      for (int msk = 1; msk < 16; msk <<= 1) m = fmaxf(m, __shfl_xor(m, msk));
      float s = 0.f;
#pragma unroll
      for (int nt = 0; nt < 4; ++nt) { vals[nt] = __expf(vals[nt] - m); s += vals[nt]; }
#pragma unroll
      for (int msk = 1; msk < 16; msk <<= 1) s += __shfl_xor(s, msk);
      float inv = 1.0f / s;
#pragma unroll
      for (int nt = 0; nt < 4; ++nt)
        ldsP[wv][(hi8 + e) * 64 + nt * 16 + lrow] = (_Float16)(vals[nt] * inv);
    }
    __syncthreads();
    v8f O[2] = {};
#pragma unroll
    for (int kh = 0; kh < 2; ++kh) {
      v16h aP = load_frag(&ldsP[wv][lrow * 64 + kh * 32 + hi8]);
#pragma unroll
      for (int n2 = 0; n2 < 2; ++n2) {
        v16h bV = load_frag(vb + (size_t)(n2 * 16 + lrow) * NPAD + kh * 32 + hi8);
        O[n2] = wmma_f16(aP, bV, O[n2]);
      }
    }
    __syncthreads();
#pragma unroll
    for (int n2 = 0; n2 < 2; ++n2)
#pragma unroll
      for (int e = 0; e < 8; ++e) {
        int M = m0 + hi8 + e;
        if (M < NTOK)
          out[((size_t)win * NTOK + M) * SDIM + head * HD + n2 * 16 + lrow] =
              (_Float16)O[n2][e];
      }
  }
}

// ---------------------------------------------------------------------------
extern "C" void kernel_launch(void* const* d_in, const int* in_sizes, int n_in,
                              void* d_out, int out_size, void* d_ws, size_t ws_size,
                              hipStream_t stream) {
  const float* x      = (const float*)d_in[0];
  const float* ln1_g  = (const float*)d_in[1];
  const float* ln1_b  = (const float*)d_in[2];
  const float* qkv_w  = (const float*)d_in[3];
  const float* qkv_b  = (const float*)d_in[4];
  const float* proj_w = (const float*)d_in[5];
  const float* proj_b = (const float*)d_in[6];
  const float* rel_b  = (const float*)d_in[7];
  const float* ln2_g  = (const float*)d_in[8];
  const float* ln2_b  = (const float*)d_in[9];
  const float* fc1_w  = (const float*)d_in[10];
  const float* fc1_b  = (const float*)d_in[11];
  const float* fc2_w  = (const float*)d_in[12];
  const float* fc2_b  = (const float*)d_in[13];
  (void)in_sizes; (void)n_in; (void)out_size; (void)ws_size;

  char* ws = (char*)d_ws;
  size_t off = 0;
  auto take = [&](size_t bytes) -> char* {
    char* p = ws + off;
    off += (bytes + 255) & ~(size_t)255;
    return p;
  };
  _Float16* wqkvT  = (_Float16*)take((size_t)576 * 192 * 2);
  _Float16* wprojT = (_Float16*)take((size_t)192 * 192 * 2);
  _Float16* wfc1T  = (_Float16*)take((size_t)768 * 192 * 2);
  _Float16* wfc2T  = (_Float16*)take((size_t)192 * 768 * 2);
  float*    bias_tab = (float*)take((size_t)HEADS * 4096 * 4);
  float*    mask_tab = (float*)take((size_t)64 * 4096 * 4);
  _Float16* h16    = (_Float16*)take((size_t)MTOT * SDIM * 2);     // LN out (reused LN2)
  size_t qkv_bytes = (size_t)3 * NWH * NPAD * HD * 2;              // Q,K,V^T (padded)
  char*  qkv_reg   = take(qkv_bytes);
  _Float16* qb = (_Float16*)qkv_reg;
  _Float16* kb = qb + (size_t)NWH * NPAD * HD;
  _Float16* vb = kb + (size_t)NWH * NPAD * HD;
  _Float16* attn_o = (_Float16*)take((size_t)MTOT * SDIM * 2);
  float*    x2     = (float*)take((size_t)MTOT * SDIM * 4);
  // MLP hidden (154 MB) aliases the dead Q/K/V + attn_o regions (189 MB)
  _Float16* mlp1 = (_Float16*)qkv_reg;

  // 1) f16 transposed weights + attention bias/mask tables
  cvt_wT<<<(192 * 576 + 255) / 256, 256, 0, stream>>>(qkv_w, wqkvT, 192, 576);
  cvt_wT<<<(192 * 192 + 255) / 256, 256, 0, stream>>>(proj_w, wprojT, 192, 192);
  cvt_wT<<<(192 * 768 + 255) / 256, 256, 0, stream>>>(fc1_w, wfc1T, 192, 768);
  cvt_wT<<<(768 * 192 + 255) / 256, 256, 0, stream>>>(fc2_w, wfc2T, 768, 192);
  build_tabs<<<16, 256, 0, stream>>>(rel_b, bias_tab, mask_tab);
  // 2) zero padded Q/K/V so token-pad lanes contribute exact zeros to WMMA
  (void)hipMemsetAsync(qkv_reg, 0, qkv_bytes, stream);
  // 3) LN1 + shift + window partition
  ln_kernel<true><<<MTOT / 8, 256, 0, stream>>>(x, ln1_g, ln1_b, h16);
  // 4) QKV GEMM (100352x192 @ 192x576) with scatter epilogue
  gemm_wmma<0><<<dim3(MTOT / 128, 576 / 64), 256, 0, stream>>>(
      h16, wqkvT, qkv_b, 192, nullptr, nullptr, nullptr, qb, kb, vb);
  // 5) windowed attention (12288 window-head pairs, 8 waves/block)
  attn_kernel<<<NWH / 8, 256, 0, stream>>>(qb, kb, vb, bias_tab, mask_tab, attn_o);
  // 6) proj GEMM + window reverse + unshift + residual -> x2 (fp32)
  gemm_wmma<1><<<dim3(MTOT / 128, 192 / 64), 256, 0, stream>>>(
      attn_o, wprojT, proj_b, 192, nullptr, x2, x, nullptr, nullptr, nullptr);
  // 7) LN2
  ln_kernel<false><<<MTOT / 8, 256, 0, stream>>>(x2, ln2_g, ln2_b, h16);
  // 8) FC1 + GELU
  gemm_wmma<2><<<dim3(MTOT / 128, 768 / 64), 256, 0, stream>>>(
      h16, wfc1T, fc1_b, 192, mlp1, nullptr, nullptr, nullptr, nullptr, nullptr);
  // 9) FC2 + residual -> d_out
  gemm_wmma<3><<<dim3(MTOT / 128, 192 / 64), 256, 0, stream>>>(
      mlp1, wfc2T, fc2_b, 768, nullptr, (float*)d_out, x2, nullptr, nullptr, nullptr);
}
